// TriangleUpdateUngated_68178310857257
// MI455X (gfx1250) — matrix operand order repaired and checked by reading, not compiled, not run on hardware
//
#include <hip/hip_runtime.h>

// ---------------------------------------------------------------------------
// TriangleUpdateUngated for MI455X (gfx1250, wave32, WMMA bf16 + TDM + async)
// B=4, L=256, E=128, H=8, D=16
// ---------------------------------------------------------------------------

typedef __attribute__((ext_vector_type(16))) __bf16 v16bf;
typedef __attribute__((ext_vector_type(8)))  __bf16 v8bf;
typedef __attribute__((ext_vector_type(8)))  float  v8f;
typedef __attribute__((ext_vector_type(4)))  float  v4f;
typedef __attribute__((ext_vector_type(4)))  unsigned u32x4;
typedef __attribute__((ext_vector_type(8)))  int      i32x8;
typedef __attribute__((ext_vector_type(4)))  int      i32x4;

static constexpr int Bb = 4;
static constexpr int Ll = 256;
static constexpr int Ee = 128;   // E
static constexpr int E2 = 256;   // 2E
static constexpr int Hh = 8;

static __device__ __forceinline__ v16bf cat8(v8bf lo, v8bf hi) {
    v16bf r;
#pragma unroll
    for (int i = 0; i < 8; ++i) { r[i] = lo[i]; r[8 + i] = hi[i]; }
    return r;
}

static __device__ __forceinline__ v8f wmma_bf16(v16bf a, v16bf b, v8f c) {
    return __builtin_amdgcn_wmma_f32_16x16x32_bf16(
        false, a, false, b, (short)0, c, false, false);
}

static __device__ __forceinline__ float wave_sum(float v) {
#pragma unroll
    for (int o = 16; o > 0; o >>= 1) v += __shfl_xor(v, o, 32);
    return v;
}
static __device__ __forceinline__ float wave_max(float v) {
#pragma unroll
    for (int o = 16; o > 0; o >>= 1) v = fmaxf(v, __shfl_xor(v, o, 32));
    return v;
}

// --- CDNA5 async global->LDS copy (ASYNCcnt path) --------------------------
static __device__ __forceinline__ void async_load_b128(const void* gsrc, void* ldst) {
    asm volatile("global_load_async_to_lds_b128 %0, %1, off"
                 :: "v"((unsigned)(size_t)ldst),
                    "v"((unsigned long long)(size_t)gsrc)
                 : "memory");
}
static __device__ __forceinline__ void wait_async0() {
    asm volatile("s_wait_asynccnt 0" ::: "memory");
}

// --- CDNA5 Tensor Data Mover: 3D bf16 tile load (TENSORcnt path) -----------
// tile = t0 (contig, elems) x t1 (stride s0 elems) x t2 (stride s1 elems)
// written contiguously to LDS.  D# packing per CDNA5 ISA (data_size=2B,
// type=2, tensor dims == tile dims).
static __device__ __forceinline__ void tdm_load_bf16_3d(
    const void* gsrc, void* ldst, unsigned t0, unsigned t1, unsigned t2,
    unsigned long long s0, unsigned long long s1) {
    unsigned long long ga = (unsigned long long)(size_t)gsrc;
    unsigned lds = (unsigned)(size_t)ldst;
    u32x4 g0 = {0, 0, 0, 0};
    g0[0] = 1u;                                    // count=1 user descriptor
    g0[1] = lds;                                   // lds_addr
    g0[2] = (unsigned)ga;                          // global_addr[31:0]
    g0[3] = (unsigned)((ga >> 32) & 0x01FFFFFFull) | (2u << 30); // [56:32]|type=2
    i32x8 g1 = {0, 0, 0, 0, 0, 0, 0, 0};
    g1[0] = (int)(1u << 16);                       // data_size = 2 bytes
    g1[1] = (int)((t0 & 0xFFFFu) << 16);           // tensor_dim0[15:0]
    g1[2] = (int)((t0 >> 16) | ((t1 & 0xFFFFu) << 16));   // dim0 hi | dim1 lo
    g1[3] = (int)((t1 >> 16) | ((t0 & 0xFFFFu) << 16));   // dim1 hi | tile_dim0
    g1[4] = (int)((t1 & 0xFFFFu) | ((t2 & 0xFFFFu) << 16)); // tile_dim1|tile_dim2
    g1[5] = (int)(unsigned)s0;                     // tensor_dim0_stride[31:0]
    g1[6] = (int)((unsigned)(s0 >> 32) | (((unsigned)s1 & 0xFFFFu) << 16));
    g1[7] = (int)(unsigned)(s1 >> 16);             // tensor_dim1_stride[47:16]
    i32x4 g2 = {(int)t2, 0, 0, 0};                 // tensor_dim2
    i32x4 g3 = {0, 0, 0, 0};
#if defined(__clang_major__) && (__clang_major__ >= 23)
    i32x8 g4 = {0, 0, 0, 0, 0, 0, 0, 0};
    __builtin_amdgcn_tensor_load_to_lds(g0, g1, g2, g3, g4, 0);
#else
    __builtin_amdgcn_tensor_load_to_lds(g0, g1, g2, g3, 0);
#endif
}

// ---------------------------------------------------------------------------
// k0: weight conversion: WVEt[n][k] (n<256 -> W_V, else W_E), WOt[n][k]=W_O[k][n]
// ---------------------------------------------------------------------------
__global__ void k0_wcvt(const float* __restrict__ W_V, const float* __restrict__ W_E,
                        const float* __restrict__ W_O,
                        __bf16* __restrict__ WVEt, __bf16* __restrict__ WOt) {
    int idx = blockIdx.x * 256 + threadIdx.x;
    if (idx < 272 * 128) {
        int n = idx >> 7, k = idx & 127;
        float w = (n < 256) ? W_V[k * 256 + n] : W_E[k * 16 + (n - 256)];
        WVEt[idx] = (__bf16)w;
    } else {
        int j2 = idx - 272 * 128;
        if (j2 < 128 * 256) {
            int n = j2 >> 8, k = j2 & 255;
            WOt[j2] = (__bf16)W_O[k * 128 + n];
        }
    }
}

// ---------------------------------------------------------------------------
// k1: fused LayerNorm + projection.  Block = 256 threads = 8 waves, 128 rows.
// e streamed non-temporally (read-once, keep L2 for k3's reuse).
// ---------------------------------------------------------------------------
__global__ __launch_bounds__(256) void k1_ln_proj(
    const float* __restrict__ e, const float* __restrict__ ln_w,
    const float* __restrict__ ln_b, const float* __restrict__ b_V,
    const float* __restrict__ b_E, const __bf16* __restrict__ WVEt,
    __bf16* __restrict__ Vws, float* __restrict__ Ep) {
    __shared__ __bf16 elds[128 * 128];   // 32 KB
    const int wave = threadIdx.x >> 5, lane = threadIdx.x & 31;
    const long row0 = (long)blockIdx.x * 128;

    const v4f lw = *((const v4f*)ln_w + lane);
    const v4f lb = *((const v4f*)ln_b + lane);
#pragma unroll 4
    for (int r = 0; r < 16; ++r) {
        const int lr = wave * 16 + r;
        v4f v = __builtin_nontemporal_load(((const v4f*)(e + (row0 + lr) * Ee)) + lane);
        float mu = wave_sum(v[0] + v[1] + v[2] + v[3]) * (1.f / 128.f);
        v4f d = v - mu;
        float var = wave_sum(d[0]*d[0] + d[1]*d[1] + d[2]*d[2] + d[3]*d[3]) * (1.f / 128.f);
        float rs = rsqrtf(var + 1e-5f);
        v4f o = d * rs * lw + lb;
        __bf16* dst = elds + lr * 128 + lane * 4;
        dst[0] = (__bf16)o[0]; dst[1] = (__bf16)o[1];
        dst[2] = (__bf16)o[2]; dst[3] = (__bf16)o[3];
    }
    __syncthreads();

    const int m = lane & 15, half = lane >> 4;
    const int m0 = wave * 16;
    const __bf16* arow = elds + (m0 + m) * 128;
    for (int t = 0; t < 17; ++t) {
        const int n = t * 16 + m;
        const __bf16* brow = WVEt + n * 128;
        v8f acc = {};
#pragma unroll
        for (int k0 = 0; k0 < 128; k0 += 32) {
            v16bf a = cat8(*(const v8bf*)(arow + k0 + half * 8),
                           *(const v8bf*)(arow + k0 + 16 + half * 8));
            v16bf b = cat8(*(const v8bf*)(brow + k0 + half * 8),
                           *(const v8bf*)(brow + k0 + 16 + half * 8));
            acc = wmma_bf16(a, b, acc);
        }
        if (n < 256) {
            const float bias = b_V[n];
#pragma unroll
            for (int vg = 0; vg < 8; ++vg) {
                long gr = row0 + m0 + vg + 8 * half;
                Vws[gr * E2 + n] = (__bf16)(acc[vg] + bias);
            }
        } else {
            const float bias = b_E[n - 256];
#pragma unroll
            for (int vg = 0; vg < 8; ++vg) {
                long gr = row0 + m0 + vg + 8 * half;
                Ep[gr * 16 + (n - 256)] = acc[vg] + bias;
            }
        }
    }
}

// ---------------------------------------------------------------------------
// k2: dual softmax -> head-packed bf16 A matrices, layout [b][h][i][k].
// ---------------------------------------------------------------------------
__global__ __launch_bounds__(256) void k2_softmax(
    const float* __restrict__ Ep, const float* __restrict__ mask,
    __bf16* __restrict__ Ain, __bf16* __restrict__ Aout) {
    const int wave = threadIdx.x >> 5, lane = threadIdx.x & 31;
    const int gw = blockIdx.x * 8 + wave;
    const int dir = gw >> 13;
    const int r = gw & 8191;
    const int b = r >> 11, i = (r >> 3) & 255, h = r & 7;

    float v[8];
    float mx = -3.4e38f;
#pragma unroll
    for (int t = 0; t < 8; ++t) {
        const int k = t * 32 + lane;
        long li, mi;
        if (dir == 0) {
            li = ((long)((b * 256 + i) * 256 + k)) * 16 + h;
            mi = (long)(b * 256 + i) * 256 + k;
        } else {
            li = ((long)((b * 256 + k) * 256 + i)) * 16 + 8 + h;
            mi = (long)(b * 256 + k) * 256 + i;
        }
        v[t] = Ep[li] + mask[mi];
        mx = fmaxf(mx, v[t]);
    }
    mx = wave_max(mx);
    float s = 0.f;
#pragma unroll
    for (int t = 0; t < 8; ++t) { v[t] = __expf(v[t] - mx); s += v[t]; }
    s = wave_sum(s);
    const float inv = 1.f / s;
    __bf16* dst = (dir == 0 ? Ain : Aout) + ((long)((b * 8 + h) * 256 + i)) * 256;
#pragma unroll
    for (int t = 0; t < 8; ++t) dst[t * 32 + lane] = (__bf16)(v[t] * inv);
}

// ---------------------------------------------------------------------------
// k3: attention einsums.  One block per (dir, b, j); wave = head h.
// V tile staged transposed in LDS; A tiles DMA'd by the Tensor Data Mover
// (3D strided tile: k(256) x i(16) x h(8), one TDM per i-tile, TENSORcnt);
// C tiles staged in LDS and written back with coalesced non-temporal b128s.
// Dynamic LDS: 64 + 64 + 4 KB.
// ---------------------------------------------------------------------------
__global__ __launch_bounds__(256) void k3_attn(
    const __bf16* __restrict__ Vws, const __bf16* __restrict__ Ain,
    const __bf16* __restrict__ Aout, __bf16* __restrict__ Va) {
    extern __shared__ char smem[];
    __bf16* Vt = (__bf16*)smem;          // [c=128][k=256]
    __bf16* Al = Vt + 128 * 256;         // [h*16+ii=128][k=256]
    __bf16* Ct = Al + 128 * 256;         // [ii=16][c=128]

    const int bid = blockIdx.x;
    const int dir = bid >> 10;
    const int bj = bid & 1023;
    const int b = bj >> 8, j = bj & 255;
    const int wave = threadIdx.x >> 5, lane = threadIdx.x & 31;

    // Stage V tile (transpose on LDS store).
    for (int q = threadIdx.x; q < 4096; q += 256) {
        const int k = q >> 4, c0 = (q & 15) << 3;
        const long src = (dir == 0)
            ? (((long)((b * 256 + j) * 256 + k)) * E2 + c0)
            : (((long)((b * 256 + k) * 256 + j)) * E2 + 128 + c0);
        v8bf v = *(const v8bf*)(Vws + src);
#pragma unroll
        for (int t2 = 0; t2 < 8; ++t2) Vt[(c0 + t2) * 256 + k] = v[t2];
    }
    const __bf16* Ab = (dir == 0) ? Ain : Aout;

    const int m = lane & 15, half = lane >> 4;
    const int h = wave;
    const __bf16* brow = Vt + (m * 8 + h) * 256;   // column d = m -> c = d*8+h

    for (int i0 = 0; i0 < 256; i0 += 16) {
        if (wave == 0) {
            // A tile for all 8 heads: contiguous k(256), i stride 256, h stride 65536
            tdm_load_bf16_3d(Ab + (long)b * 8 * 65536 + (long)i0 * 256, Al,
                             256u, 16u, 8u, 256ull, 65536ull);
            __builtin_amdgcn_s_wait_tensorcnt(0);
        }
        __syncthreads();

        const __bf16* arow = Al + (h * 16 + m) * 256;
        v8f acc = {};
#pragma unroll
        for (int k0 = 0; k0 < 256; k0 += 32) {
            v16bf a = cat8(*(const v8bf*)(arow + k0 + half * 8),
                           *(const v8bf*)(arow + k0 + 16 + half * 8));
            v16bf bm = cat8(*(const v8bf*)(brow + k0 + half * 8),
                            *(const v8bf*)(brow + k0 + 16 + half * 8));
            acc = wmma_bf16(a, bm, acc);
        }
        // Stage C tile (16 i x 128 c, all heads interleave c = d*8+h).
#pragma unroll
        for (int vg = 0; vg < 8; ++vg)
            Ct[(vg + 8 * half) * 128 + m * 8 + h] = (__bf16)acc[vg];
        __syncthreads();
        {
            const int q = threadIdx.x;               // 256 threads = 256 x v8bf
            const int ii = q >> 4, c0 = (q & 15) << 3;
            v8bf vv = *(const v8bf*)(Ct + ii * 128 + c0);
            __bf16* dst = Va + ((long)((b * 256 + (i0 + ii)) * 256 + j)) * E2
                          + (dir ? 128 : 0) + c0;
            __builtin_nontemporal_store(vv, (v8bf*)dst);
        }
        __syncthreads();
    }
}

// ---------------------------------------------------------------------------
// k3b: out = Va @ W_O + b_O.  W staged via async global->LDS (ASYNCcnt);
// Va read non-temporally, out written non-temporally.
// ---------------------------------------------------------------------------
__global__ __launch_bounds__(256) void k3b_outproj(
    const __bf16* __restrict__ Va, const __bf16* __restrict__ WOt,
    const float* __restrict__ b_O, float* __restrict__ out) {
    extern __shared__ char smem[];
    __bf16* Wl = (__bf16*)smem;  // [n=128][k=256], 64 KB
    for (int q = threadIdx.x; q < 4096; q += 256)
        async_load_b128(WOt + (size_t)q * 8, Wl + (size_t)q * 8);
    wait_async0();
    __syncthreads();

    const int wave = threadIdx.x >> 5, lane = threadIdx.x & 31;
    const int m = lane & 15, half = lane >> 4;
    const long row0 = (long)blockIdx.x * 128 + wave * 16;
    const __bf16* arow = Va + (row0 + m) * E2;

    for (int t = 0; t < 8; ++t) {
        const int n = t * 16 + m;
        const __bf16* brow = Wl + n * 256;
        v8f acc = {};
#pragma unroll
        for (int k0 = 0; k0 < 256; k0 += 32) {
            v8bf alo = __builtin_nontemporal_load((const v8bf*)(arow + k0 + half * 8));
            v8bf ahi = __builtin_nontemporal_load((const v8bf*)(arow + k0 + 16 + half * 8));
            v16bf a = cat8(alo, ahi);
            v16bf bm = cat8(*(const v8bf*)(brow + k0 + half * 8),
                            *(const v8bf*)(brow + k0 + 16 + half * 8));
            acc = wmma_bf16(a, bm, acc);
        }
        const float bias = b_O[n];
#pragma unroll
        for (int vg = 0; vg < 8; ++vg) {
            const long r = row0 + vg + 8 * half;
            __builtin_nontemporal_store(acc[vg] + bias, &out[r * 128 + n]);
        }
    }
}

// ---------------------------------------------------------------------------
// Launch
// ---------------------------------------------------------------------------
extern "C" void kernel_launch(void* const* d_in, const int* in_sizes, int n_in,
                              void* d_out, int out_size, void* d_ws, size_t ws_size,
                              hipStream_t stream) {
    const float* e    = (const float*)d_in[0];
    const float* mask = (const float*)d_in[1];
    const float* ln_w = (const float*)d_in[2];
    const float* ln_b = (const float*)d_in[3];
    const float* W_V  = (const float*)d_in[4];
    const float* b_V  = (const float*)d_in[5];
    const float* W_E  = (const float*)d_in[6];
    const float* b_E  = (const float*)d_in[7];
    const float* W_O  = (const float*)d_in[8];
    const float* b_O  = (const float*)d_in[9];
    float* out = (float*)d_out;

    char* ws = (char*)d_ws;
    const size_t NROWS = (size_t)Bb * Ll * Ll;             // 262144
    __bf16* Vws  = (__bf16*)(ws);                          // 134 MB
    __bf16* Vaws = (__bf16*)(ws + NROWS * E2 * 2);         // 134 MB
    float*  Epws = (float*)(ws + 2 * NROWS * E2 * 2);      // 16 MB
    __bf16* Ainw = (__bf16*)(ws + 2 * NROWS * E2 * 2 + NROWS * 16 * 4);
    __bf16* Aoutw= Ainw + (size_t)Bb * Hh * Ll * Ll;
    __bf16* WVEt = Aoutw + (size_t)Bb * Hh * Ll * Ll;
    __bf16* WOt  = WVEt + 272 * 128;

    k0_wcvt<<<(272 * 128 + 128 * 256) / 256, 256, 0, stream>>>(W_V, W_E, W_O, WVEt, WOt);
    k1_ln_proj<<<(int)(NROWS / 128), 256, 0, stream>>>(e, ln_w, ln_b, b_V, b_E, WVEt, Vws, Epws);
    k2_softmax<<<2 * Bb * Ll * Hh / 8, 256, 0, stream>>>(Epws, mask, Ainw, Aoutw);
    k3_attn<<<2 * Bb * Ll, 256, (128 * 256 + 128 * 256 + 16 * 128) * 2, stream>>>(
        Vws, Ainw, Aoutw, Vaws);
    k3b_outproj<<<(int)(NROWS / 128), 256, 128 * 256 * 2, stream>>>(Vaws, WOt, b_O, out);
}
